// MihGNNEmbedding10_4947802325014
// MI455X (gfx1250) — matrix operand order repaired and checked by reading, not compiled, not run on hardware
//
#include <hip/hip_runtime.h>
#include <hip/hip_bf16.h>
#include <stdint.h>

#define N_NODES 50000
#define DIM     128
#define KNBR    16
#define NPAIR   100000

typedef __attribute__((ext_vector_type(16))) __bf16       v16bf;
typedef __attribute__((ext_vector_type(8)))  float        v8f;
typedef __attribute__((ext_vector_type(8)))  unsigned int v8u;

// ---------------------------------------------------------------------------
// Zero the scalar loss accumulator (harness poisons d_out with 0xAA).
// ---------------------------------------------------------------------------
__global__ void zero_out_kernel(float* out) {
    if (threadIdx.x == 0 && blockIdx.x == 0) out[0] = 0.0f;
}

// ---------------------------------------------------------------------------
// Repack W1 (fp32 [256,128]) into bf16 pair-packed dwords, already in the
// WMMA B-fragment-transposed order:
//   W1p[ ((c*2 + half)*128 + col) * 8 + j ] = { W1[2k][col], W1[2k+1][col] }
//   with k = c*16 + half*8 + j,  col = n*16 + m.
// This makes (a) the per-block LDS staging a LINEAR 64KB copy (async b128
// friendly) and (b) each lane's B-fragment 8 CONTIGUOUS LDS dwords
// (2 x ds_load_b128).
// ---------------------------------------------------------------------------
__global__ void pack_w1_kernel(const float* __restrict__ W1,
                               uint32_t* __restrict__ W1p) {
    int idx = blockIdx.x * blockDim.x + threadIdx.x;
    if (idx >= 128 * 128) return;
    int j    = idx & 7;
    int col  = (idx >> 3) & 127;
    int ch   = idx >> 10;            // c*2 + half, 0..15
    int c    = ch >> 1;
    int half = ch & 1;
    int k    = c * 16 + half * 8 + j;   // K-pair index (rows 2k, 2k+1 of W1)
    __bf16 lo = (__bf16)W1[(size_t)(2 * k)     * DIM + col];
    __bf16 hi = (__bf16)W1[(size_t)(2 * k + 1) * DIM + col];
    W1p[idx] = (uint32_t)__builtin_bit_cast(uint16_t, lo) |
               ((uint32_t)__builtin_bit_cast(uint16_t, hi) << 16);
}

// ---------------------------------------------------------------------------
// One mean-aggregation layer: dst[n] = mean_k src[nbr[n,k]]
// One block per node, 128 threads = one fp32 element each; 512B coalesced
// bursts per neighbor row, L2-resident table (25.6MB << 192MB L2).
// ---------------------------------------------------------------------------
__global__ void aggregate_kernel(const float* __restrict__ src,
                                 const int* __restrict__ nbr,
                                 float* __restrict__ dst) {
    int n = blockIdx.x;
    int d = threadIdx.x;
    const int* nb = nbr + (size_t)n * KNBR;
    float s = 0.0f;
    #pragma unroll
    for (int k = 0; k < KNBR; ++k)
        s += src[(size_t)nb[k] * DIM + d];
    dst[(size_t)n * DIM + d] = s * (1.0f / KNBR);
}

// ---------------------------------------------------------------------------
// Pair MLP + loss. 8 waves/block, one 16-pair tile per wave.
// W1 (packed bf16, 64KB) is staged into LDS once per block with
// GLOBAL_LOAD_ASYNC_TO_LDS_B128 (ASYNCcnt path), then every wave's
// B-fragments come from ds_load_b128 — cutting redundant L2 reads 8x.
// GEMM: v_wmma_f32_16x16x32_bf16, 8 K-chunks x 8 N-tiles = 64 WMMAs/wave.
// ---------------------------------------------------------------------------
__global__ void __launch_bounds__(256)
pair_mlp_loss_kernel(const int* __restrict__ pairs,
                     const int* __restrict__ labels,
                     const float* __restrict__ H,
                     const uint32_t* __restrict__ W1p,
                     const float* __restrict__ b1,
                     const float* __restrict__ W2,
                     const float* __restrict__ b2,
                     float* __restrict__ out) {
    __shared__ __attribute__((aligned(16))) uint32_t smemW1[16384]; // 64KB

    const int lane = threadIdx.x & 31;
    const int wave = threadIdx.x >> 5;
    const int tile = blockIdx.x * 8 + wave;

    // ---- Stage packed W1 into LDS: linear 64KB async copy, 16 x b128/thread.
    {
        uint32_t ldsBase = (uint32_t)(uintptr_t)(&smemW1[0]); // LDS offset = low 32b
        #pragma unroll
        for (int i = 0; i < 16; ++i) {
            uint32_t byteOff = (uint32_t)(threadIdx.x + i * 256) * 16u;
            uint32_t ldsAddr = ldsBase + byteOff;
            asm volatile("global_load_async_to_lds_b128 %0, %1, %2"
                         :: "v"(ldsAddr), "v"(byteOff), "s"(W1p)
                         : "memory");
        }
        asm volatile("s_wait_asynccnt 0" ::: "memory");
    }
    __syncthreads();

    if (tile * 16 >= NPAIR) return;           // whole-wave uniform exit

    const int m        = lane & 15;           // row within tile / B column
    const int half     = lane >> 4;           // lane half selects K sub-block
    const int tileBase = tile * 16;

    // Pair indices for row m of this tile (same for both lane halves).
    const int p    = tileBase + m;
    const int sIdx = pairs[2 * p];
    const int dIdx = pairs[2 * p + 1];
    const float* rs = H + (size_t)sIdx * DIM;
    const float* rd = H + (size_t)dIdx * DIM;

    v8f acc[8] = {};                          // 8 output N-tiles, f32 16x16

    #pragma unroll
    for (int c = 0; c < 8; ++c) {             // K chunks of 32 (256 total)
        // cols 0..127 = src row, 128..255 = dst row (concat never straddles)
        const float* row = (c < 4) ? (rs + c * 32) : (rd + (c - 4) * 32);

        // A fragment: contiguous k = 16*half + s, same order as the W1p B packing.
        v16bf a;
        #pragma unroll
        for (int j = 0; j < 8; ++j) {
            int kl = 2 * j + half * 16;
            a[2 * j]     = (__bf16)row[kl];
            a[2 * j + 1] = (__bf16)row[kl + 1];
        }

        const uint32_t* chunkBase = smemW1 + (size_t)(c * 2 + half) * 128 * 8;

        #pragma unroll
        for (int n = 0; n < 8; ++n) {         // output column tiles
            // B fragment: 8 contiguous LDS dwords -> 2 x ds_load_b128.
            const uint32_t* frag = chunkBase + (size_t)(n * 16 + m) * 8;
            v8u bw;
            #pragma unroll
            for (int j = 0; j < 8; ++j) bw[j] = frag[j];
            v16bf b = __builtin_bit_cast(v16bf, bw);

            acc[n] = __builtin_amdgcn_wmma_f32_16x16x32_bf16(
                false, a, false, b, (short)0, acc[n], false, false);
        }
    }

    // C layout: VGPR r, lane -> row M = r + 8*half, col N = n*16 + m.
    // Bias + ReLU, then partial dot with W2 columns (2 outputs).
    float s0[8], s1[8];
    #pragma unroll
    for (int r = 0; r < 8; ++r) { s0[r] = 0.0f; s1[r] = 0.0f; }

    #pragma unroll
    for (int n = 0; n < 8; ++n) {
        int col    = n * 16 + m;
        float bias = b1[col];
        float w20  = W2[col * 2 + 0];
        float w21  = W2[col * 2 + 1];
        #pragma unroll
        for (int r = 0; r < 8; ++r) {
            float h = acc[n][r] + bias;
            h = fmaxf(h, 0.0f);
            s0[r] = fmaf(h, w20, s0[r]);
            s1[r] = fmaf(h, w21, s1[r]);
        }
    }

    // Reduce across the 16 lanes of each half (xor masks 1/2/4/8 stay in-half).
    #pragma unroll
    for (int off = 8; off >= 1; off >>= 1) {
        #pragma unroll
        for (int r = 0; r < 8; ++r) {
            s0[r] += __shfl_xor(s0[r], off, 32);
            s1[r] += __shfl_xor(s1[r], off, 32);
        }
    }

    if (m == 0) {                             // lanes 0 and 16 finalize
        float bb0 = b2[0], bb1 = b2[1];
        float local = 0.0f;
        #pragma unroll
        for (int r = 0; r < 8; ++r) {
            int pidx = tileBase + r + half * 8;
            float l0 = s0[r] + bb0;
            float l1 = s1[r] + bb1;
            // softmax
            float mx = fmaxf(l0, l1);
            float e0 = __expf(l0 - mx), e1 = __expf(l1 - mx);
            float Z  = e0 + e1;
            float p0 = e0 / Z, p1 = e1 / Z;
            // log_softmax of the probabilities (reference double-softmax)
            float mx2 = fmaxf(p0, p1);
            float lz  = logf(__expf(p0 - mx2) + __expf(p1 - mx2)) + mx2;
            float py  = labels[pidx] ? p1 : p0;
            local += (lz - py);               // -logp[label]
        }
        atomicAdd(out, local * (1.0f / NPAIR));
    }
}

// ---------------------------------------------------------------------------
extern "C" void kernel_launch(void* const* d_in, const int* in_sizes, int n_in,
                              void* d_out, int out_size, void* d_ws, size_t ws_size,
                              hipStream_t stream) {
    const int*   pairs     = (const int*)d_in[0];
    const int*   labels    = (const int*)d_in[1];
    const int*   neighbors = (const int*)d_in[2];
    const float* embedding = (const float*)d_in[3];
    const float* W1        = (const float*)d_in[4];
    const float* b1        = (const float*)d_in[5];
    const float* W2        = (const float*)d_in[6];
    const float* b2        = (const float*)d_in[7];
    float* out = (float*)d_out;

    // Workspace layout: H1 (25.6MB) | H2 (25.6MB) | W1 packed bf16 (64KB)
    char* ws = (char*)d_ws;
    float*    H1  = (float*)ws;
    float*    H2  = (float*)(ws + (size_t)N_NODES * DIM * sizeof(float));
    uint32_t* W1p = (uint32_t*)(ws + 2 * (size_t)N_NODES * DIM * sizeof(float));

    zero_out_kernel<<<1, 32, 0, stream>>>(out);
    pack_w1_kernel<<<(128 * 128 + 255) / 256, 256, 0, stream>>>(W1, W1p);
    aggregate_kernel<<<N_NODES, DIM, 0, stream>>>(embedding, neighbors, H1);
    aggregate_kernel<<<N_NODES, DIM, 0, stream>>>(H1, neighbors, H2);

    int tiles  = NPAIR / 16;            // 6250 full tiles
    int blocks = (tiles + 7) / 8;       // 8 waves (tiles) per 256-thread block
    pair_mlp_loss_kernel<<<blocks, 256, 0, stream>>>(
        pairs, labels, H2, W1p, b1, W2, b2, out);
}
